// Attention_PC_17901423690230
// MI455X (gfx1250) — matrix operand order repaired and checked by reading, not compiled, not run on hardware
//
#include <hip/hip_runtime.h>

// CDNA5 / gfx1250: wave32, WMMA 16x16x32 bf16 -> f32 accumulate.
typedef __attribute__((ext_vector_type(16))) __bf16 v16bf;
typedef __attribute__((ext_vector_type(8)))  float  v8f;
typedef __attribute__((ext_vector_type(4)))  float  v4f;   // native vector for NT builtins
typedef __attribute__((ext_vector_type(2)))  float  v2f;

__device__ __forceinline__ float fast_exp(float x) {
  return __builtin_amdgcn_exp2f(x * 1.4426950408889634f);   // v_exp_f32
}
__device__ __forceinline__ float fast_tanh(float x) {
  // tanh(x) = 1 - 2/(1 + exp(2x)); saturates correctly at +-1
  float e = __builtin_amdgcn_exp2f(x * 2.8853900817779268f);
  return 1.0f - 2.0f * __builtin_amdgcn_rcpf(e + 1.0f);
}

// Pack 16 fp32 (4x v4f, already in A-fragment K order) into a bf16 A fragment.
__device__ __forceinline__ v16bf pack_bf16(const v4f* q) {
  v16bf a;
#pragma unroll
  for (int j = 0; j < 4; ++j) {
    a[4 * j + 0] = (__bf16)q[j].x;
    a[4 * j + 1] = (__bf16)q[j].y;
    a[4 * j + 2] = (__bf16)q[j].z;
    a[4 * j + 3] = (__bf16)q[j].w;
  }
  return a;
}

// Per 16-row tile per wave:
//   h1 = x1_tile[16x128] @ W1[128x64], h2 = x2_tile @ W1  (WMMA bf16, f32 acc)
//   s  = W2 . leaky(h + b1)        (b2 cancels in the 2-way softmax)
//   beta = softmax([s1,s2]);  Z = tanh([beta1*x1, beta2*x2])
// B fragments are re-read from LDS at each WMMA to stay under 256 VGPRs
// (4 waves/SIMD instead of 2, no s_set_vgpr_msb tax).
__global__ __launch_bounds__(256, 4) void attn_pc_wmma(
    const float* __restrict__ x1, const float* __restrict__ x2,
    const float* __restrict__ W1, const float* __restrict__ b1,
    const float* __restrict__ W2,
    float* __restrict__ outZ, float* __restrict__ outBeta,
    int N, int n_tiles)
{
  // W1 transposed to bf16: Wt[n][k] = W1[k*64+n]; B-fragment reads are contiguous.
  __shared__ __align__(32) __bf16 Wt[64][128];     // 16 KB
  __shared__ float sbuf[8][2][16];                 // per-wave score exchange

  const int tid  = threadIdx.x;
  const int lane = tid & 31;
  const int wave = tid >> 5;
  const int half = lane >> 4;   // which 16-lane half
  const int l15  = lane & 15;
  const int q0   = half * 16;   // A-frag K offset within a 32-K step (contiguous, matches B)
  const int kb   = half * 16;   // B-frag K offset within a 32-K step

  // Cooperative W1 fp32 -> bf16 transpose into LDS (once per block).
  for (int idx = tid; idx < 128 * 64; idx += 256) {
    int k = idx >> 6, n = idx & 63;
    Wt[n][k] = (__bf16)W1[idx];
  }
  __syncthreads();

  // Per-lane second-layer weights / bias for columns n = t*16 + l15.
  float w2v[4], b1v[4];
#pragma unroll
  for (int t = 0; t < 4; ++t) {
    w2v[t] = W2[t * 16 + l15];
    b1v[t] = b1[t * 16 + l15];
  }

  const int wid     = blockIdx.x * 8 + wave;  // 8 waves / 256-thread block
  const int wstride = gridDim.x * 8;

  for (int tix = wid; tix < n_tiles; tix += wstride) {
    const int  rraw = tix * 16 + l15;
    const long row  = (rraw < N) ? rraw : (N - 1);   // clamp loads; stores are guarded
    const float* p1 = x1 + row * 128;
    const float* p2 = x2 + row * 128;

    // Prefetch next tile's rows (global_prefetch_b8, speculative).
    if (tix + wstride < n_tiles) {
      const long nrow = (long)(tix + wstride) * 16 + l15;
      __builtin_prefetch(x1 + nrow * 128, 0, 0);
      __builtin_prefetch(x2 + nrow * 128, 0, 0);
    }

    const v8f vzero = {0.f, 0.f, 0.f, 0.f, 0.f, 0.f, 0.f, 0.f};
    v8f acc1[4], acc2[4];
#pragma unroll
    for (int t = 0; t < 4; ++t) { acc1[t] = vzero; acc2[t] = vzero; }

    // fp32 x copies stay live for the tanh output phase.
    v4f xs1[4][4], xs2[4][4];

#pragma unroll
    for (int ks = 0; ks < 4; ++ks) {
      const int cb = ks * 32 + q0;
      // Streaming data: non-temporal loads, don't churn L2.
      xs1[ks][0] = __builtin_nontemporal_load((const v4f*)(p1 + cb));
      xs1[ks][1] = __builtin_nontemporal_load((const v4f*)(p1 + cb + 4));
      xs1[ks][2] = __builtin_nontemporal_load((const v4f*)(p1 + cb + 8));
      xs1[ks][3] = __builtin_nontemporal_load((const v4f*)(p1 + cb + 12));
      xs2[ks][0] = __builtin_nontemporal_load((const v4f*)(p2 + cb));
      xs2[ks][1] = __builtin_nontemporal_load((const v4f*)(p2 + cb + 4));
      xs2[ks][2] = __builtin_nontemporal_load((const v4f*)(p2 + cb + 8));
      xs2[ks][3] = __builtin_nontemporal_load((const v4f*)(p2 + cb + 12));
      const v16bf a1 = pack_bf16(xs1[ks]);
      const v16bf a2 = pack_bf16(xs2[ks]);

#pragma unroll
      for (int t = 0; t < 4; ++t) {
        // B fragment straight from LDS (2x ds_load_b128), transient in VGPRs.
        const v16bf bf = *(const v16bf*)&Wt[t * 16 + l15][ks * 32 + kb];
        acc1[t] = __builtin_amdgcn_wmma_f32_16x16x32_bf16(
            false, a1, false, bf, (short)0, acc1[t], false, false);
        acc2[t] = __builtin_amdgcn_wmma_f32_16x16x32_bf16(
            false, a2, false, bf, (short)0, acc2[t], false, false);
      }
    }

    // Scores: C/D layout -> VGPR r holds row M=r+8*half, col N=t*16+lane%16.
#pragma unroll
    for (int r = 0; r < 8; ++r) {
      float s1 = 0.f, s2 = 0.f;
#pragma unroll
      for (int t = 0; t < 4; ++t) {
        float v = acc1[t][r] + b1v[t];
        v = (v > 0.f) ? v : 0.01f * v;          // leaky_relu(0.01)
        s1 += v * w2v[t];
        float u = acc2[t][r] + b1v[t];
        u = (u > 0.f) ? u : 0.01f * u;
        s2 += u * w2v[t];
      }
      // reduce over the 16 columns held within each lane-half
#pragma unroll
      for (int m = 1; m < 16; m <<= 1) {
        s1 += __shfl_xor(s1, m, 32);
        s2 += __shfl_xor(s2, m, 32);
      }
      if (l15 == 0) {
        sbuf[wave][0][half * 8 + r] = s1;
        sbuf[wave][1][half * 8 + r] = s2;
      }
    }
    // Wave-local LDS ordering: CDNA5 split counter wait (stores -> loads).
    asm volatile("s_wait_dscnt 0" ::: "memory");
    const float sc1 = sbuf[wave][0][l15];
    const float sc2 = sbuf[wave][1][l15];

    const float mx  = fmaxf(sc1, sc2);
    const float e1  = fast_exp(sc1 - mx);
    const float e2  = fast_exp(sc2 - mx);
    const float inv = __builtin_amdgcn_rcpf(e1 + e2);
    const float be1 = e1 * inv;
    const float be2 = e2 * inv;

    if (rraw < N) {
      float* zr = outZ + (size_t)rraw * 256;
#pragma unroll
      for (int ks = 0; ks < 4; ++ks) {
        const int cb = ks * 32 + q0;
#pragma unroll
        for (int j = 0; j < 4; ++j) {
          const int co = cb + j * 4;
          v4f a = xs1[ks][j], oa;
          oa.x = fast_tanh(be1 * a.x); oa.y = fast_tanh(be1 * a.y);
          oa.z = fast_tanh(be1 * a.z); oa.w = fast_tanh(be1 * a.w);
          __builtin_nontemporal_store(oa, (v4f*)(zr + co));
          v4f b = xs2[ks][j], ob;
          ob.x = fast_tanh(be2 * b.x); ob.y = fast_tanh(be2 * b.y);
          ob.z = fast_tanh(be2 * b.z); ob.w = fast_tanh(be2 * b.w);
          __builtin_nontemporal_store(ob, (v4f*)(zr + 128 + co));
        }
      }
      if (half == 0) {
        v2f bv; bv.x = be1; bv.y = be2;
        *(v2f*)(outBeta + (size_t)rraw * 2) = bv;
      }
    }
  }
}

extern "C" void kernel_launch(void* const* d_in, const int* in_sizes, int n_in,
                              void* d_out, int out_size, void* d_ws, size_t ws_size,
                              hipStream_t stream) {
  const float* x1 = (const float*)d_in[0];
  const float* x2 = (const float*)d_in[1];
  const float* W1 = (const float*)d_in[2];
  const float* b1 = (const float*)d_in[3];
  const float* W2 = (const float*)d_in[4];
  // d_in[5] = b2: adds equally to both scores -> cancels in 2-way softmax.

  const int N       = in_sizes[0] / 128;
  const int n_tiles = (N + 15) / 16;

  float* outZ    = (float*)d_out;                 // N * 256
  float* outBeta = outZ + (size_t)N * 256;        // N * 2

  int blocks = (n_tiles + 7) / 8;                 // 8 waves per block
  if (blocks > 1024) blocks = 1024;               // grid-stride amortizes W1 setup
  if (blocks < 1) blocks = 1;

  attn_pc_wmma<<<dim3(blocks), dim3(256), 0, stream>>>(
      x1, x2, W1, b1, W2, outZ, outBeta, N, n_tiles);
}
